// VSSBlock_61400852464214
// MI455X (gfx1250) — matrix operand order repaired and checked by reading, not compiled, not run on hardware
//
#include <hip/hip_runtime.h>

// ---------------------------------------------------------------------------
// Problem constants (from reference)
// ---------------------------------------------------------------------------
#define BB  4
#define HH  64
#define WWI 64
#define CC  96
#define LLEN (HH*WWI)          // 4096
#define MM  (BB*LLEN)          // 16384 rows (b,l)
#define DE  192                // 2*C0
#define D2  384                // 2*DE
#define KDIR 4
#define NSS 16
#define RANK 6
#define CDBL (RANK+2*NSS)      // 38
#define CDBL_PAD 48            // N=38 padded to 3 tiles
#define RANK_PAD 32            // K=6 padded to one 32-chunk
#define CH2 (CC/2)             // 48
#define CH2_PAD 64             // K=48 padded to two 32-chunks

typedef __attribute__((ext_vector_type(16))) _Float16 v16h;
typedef __attribute__((ext_vector_type(8)))  _Float16 v8h;
typedef __attribute__((ext_vector_type(8)))  float    v8f;

#define ACT_NONE 0
#define ACT_SOFTPLUS 1
#define ACT_GELU 2

// ---------------------------------------------------------------------------
// Device helpers
// ---------------------------------------------------------------------------
__device__ __forceinline__ float geluf(float x) {
    return 0.5f * x * (1.0f + erff(x * 0.70710678118654752f));
}
__device__ __forceinline__ float softplusf(float x) {
    return (x > 20.0f) ? x : log1pf(expf(x));
}
__device__ __forceinline__ float sigmoidf(float x) {
    return 1.0f / (1.0f + expf(-x));
}
__device__ __forceinline__ float siluf(float x) {
    return x * sigmoidf(x);
}

// ---------------------------------------------------------------------------
// Generic WMMA GEMM:  C[m,n] = act( sum_k A[m,k]*W[n,k] + bias[n] ) + resid[m,n]
// Compile-time K (multiple of 32); operands zero-padded so the inner loop has
// NO bounds checks: per lane the A fragment is two contiguous 16B runs, the B
// fragment two contiguous 16B runs -> 4x global_load_b128 + 1x v_wmma per
// 16x16x32 chunk, fully unrolled over K.
//   A: f16 [Mrows x lda] (M multiple of 16, lda multiple of 8, zero-padded K)
//   W: f16 [>=Ntiles*16 x ldb] row-major [out,in], zero-padded rows/cols
//   C: f32 [Mrows x ldc]
// ---------------------------------------------------------------------------
template <int KD>
__global__ __launch_bounds__(256) void gemm_wmma_kernel(
    const _Float16* __restrict__ A, const _Float16* __restrict__ Wt,
    float* __restrict__ Cout, int Mrows, int Ncols,
    int lda, int ldb, int ldc,
    const float* __restrict__ bias, const float* __restrict__ resid,
    int ldr, int act)
{
    const int lane = threadIdx.x & 31;
    const int wid  = blockIdx.x * (blockDim.x >> 5) + (threadIdx.x >> 5);
    const int Mt = Mrows >> 4;
    const int Nt = (Ncols + 15) >> 4;
    if (wid >= Mt * Nt) return;
    const int mt = wid % Mt;
    const int nt = wid / Mt;

    const int hf = lane >> 4;       // half-wave selector
    const int lm = lane & 15;

    const _Float16* Arow = A  + (size_t)((mt << 4) + lm) * lda;   // lane's A row
    const _Float16* Wrow = Wt + (size_t)((nt << 4) + lm) * ldb;   // lane's W row

    v8f acc = {};
#pragma unroll
    for (int k0 = 0; k0 < KD; k0 += 32) {
        // A fragment (ISA 16-bit A 16x32): elements 0..7 = K k0+8*hf+0..7,
        // elements 8..15 = K k0+16+8*hf+0..7 -> two 16B contiguous loads.
        v8h alo = *(const v8h*)(Arow + k0 + (hf << 3));
        v8h ahi = *(const v8h*)(Arow + k0 + 16 + (hf << 3));
        v16h afrag = __builtin_shufflevector(alo, ahi,
            0, 1, 2, 3, 4, 5, 6, 7, 8, 9, 10, 11, 12, 13, 14, 15);
        // B fragment: same K order as A -> two 16B contiguous loads.
        v8h blo = *(const v8h*)(Wrow + k0 + (hf << 3));
        v8h bhi = *(const v8h*)(Wrow + k0 + 16 + (hf << 3));
        v16h bfrag = __builtin_shufflevector(blo, bhi,
            0, 1, 2, 3, 4, 5, 6, 7, 8, 9, 10, 11, 12, 13, 14, 15);
        acc = __builtin_amdgcn_wmma_f32_16x16x32_f16(
                  false, afrag, false, bfrag, (short)0, acc, false, false);
    }

    // C/D layout: VGPR v -> row 8*hf + v, col lane%16
    const int no = (nt << 4) + lm;
    if (no < Ncols) {
        float bv = bias ? bias[no] : 0.0f;
#pragma unroll
        for (int v = 0; v < 8; ++v) {
            const int mo = (mt << 4) + (hf << 3) + v;
            float val = acc[v] + bv;
            if (act == ACT_SOFTPLUS) val = softplusf(val);
            else if (act == ACT_GELU) val = geluf(val);
            if (resid) val += resid[(size_t)mo * ldr + no];
            Cout[(size_t)mo * ldc + no] = val;
        }
    }
}

// ---------------------------------------------------------------------------
// Wave-per-row LayerNorm (optionally elementwise-multiplied, f32/f16 outputs)
// ---------------------------------------------------------------------------
__global__ __launch_bounds__(256) void ln_kernel(
    const float* __restrict__ x, int ldx, int coff,
    const float* __restrict__ g, const float* __restrict__ bta,
    const float* __restrict__ mul,
    float* __restrict__ o32, _Float16* __restrict__ o16,
    int rows, int C, float eps)
{
    const int lane = threadIdx.x & 31;
    const int row  = blockIdx.x * (blockDim.x >> 5) + (threadIdx.x >> 5);
    if (row >= rows) return;
    const float* xr = x + (size_t)row * ldx + coff;
    float s = 0.0f, s2 = 0.0f;
    for (int c = lane; c < C; c += 32) { float v = xr[c]; s += v; s2 += v * v; }
    for (int o = 16; o > 0; o >>= 1) {
        s  += __shfl_xor(s,  o, 32);
        s2 += __shfl_xor(s2, o, 32);
    }
    const float mu = s / (float)C;
    const float var = s2 / (float)C - mu * mu;
    const float r = rsqrtf(var + eps);
    for (int c = lane; c < C; c += 32) {
        float v = (xr[c] - mu) * r * g[c] + bta[c];
        if (mul) v *= mul[(size_t)row * C + c];
        if (o32) o32[(size_t)row * C + c] = v;
        if (o16) o16[(size_t)row * C + c] = (_Float16)v;
    }
}

// ---------------------------------------------------------------------------
// f32 -> f16 conversion (flat)
// ---------------------------------------------------------------------------
__global__ void cvt_f32_f16_kernel(const float* __restrict__ s,
                                   _Float16* __restrict__ d, int n)
{
    int i = blockIdx.x * blockDim.x + threadIdx.x;
    if (i < n) d[i] = (_Float16)s[i];
}

// ---------------------------------------------------------------------------
// f32 -> f16 with zero padding:
// dst[r*ld_dst + c] = (r<rows_src && c<cols_src) ? src[r*ld_src + coff + c] : 0
// launched over rows_pad*ld_dst elements.
// ---------------------------------------------------------------------------
__global__ void cvt_pad_kernel(const float* __restrict__ src, int ld_src, int coff,
                               _Float16* __restrict__ dst, int ld_dst,
                               int rows_src, int cols_src, int total)
{
    int i = blockIdx.x * blockDim.x + threadIdx.x;
    if (i >= total) return;
    const int c = i % ld_dst;
    const int r = i / ld_dst;
    _Float16 v = (_Float16)0.0f;
    if (r < rows_src && c < cols_src)
        v = (_Float16)src[(size_t)r * ld_src + coff + c];
    dst[i] = v;
}

// ---------------------------------------------------------------------------
// Split in_proj output: xx -> NCHW f32, z -> silu(z) [M,DE]
// ---------------------------------------------------------------------------
__global__ void split_inproj_kernel(const float* __restrict__ xz,
                                    float* __restrict__ xx_nchw,
                                    float* __restrict__ z)
{
    int idx = blockIdx.x * blockDim.x + threadIdx.x;   // M*DE
    if (idx >= MM * DE) return;
    const int d = idx % DE;
    const int m = idx / DE;
    const int b = m / LLEN;
    const int l = m % LLEN;
    xx_nchw[((size_t)b * DE + d) * LLEN + l] = xz[(size_t)m * D2 + d];
    const float zz = xz[(size_t)m * D2 + DE + d];
    z[idx] = siluf(zz);
}

// ---------------------------------------------------------------------------
// Depthwise 3x3 + bias + SiLU on xx (NCHW), fan out 4 scan directions as f16
// xs layout: [k][b][l][d] (d contiguous) for coalesced scan + GEMM-A use.
// ---------------------------------------------------------------------------
__global__ void dwconv_xs_kernel(const float* __restrict__ xx,
                                 const float* __restrict__ w,
                                 const float* __restrict__ bias,
                                 _Float16* __restrict__ xs)
{
    int idx = blockIdx.x * blockDim.x + threadIdx.x;   // B*DE*L
    if (idx >= BB * DE * LLEN) return;
    const int l = idx % LLEN;
    const int d = (idx / LLEN) % DE;
    const int b = idx / (LLEN * DE);
    const int h = l / WWI, ww = l % WWI;
    const float* xin = xx + ((size_t)b * DE + d) * LLEN;
    float s = bias[d];
#pragma unroll
    for (int ky = 0; ky < 3; ++ky) {
        const int hy = h + ky - 1;
        if (hy < 0 || hy >= HH) continue;
#pragma unroll
        for (int kx = 0; kx < 3; ++kx) {
            const int wx = ww + kx - 1;
            if (wx < 0 || wx >= WWI) continue;
            s += w[d * 9 + ky * 3 + kx] * xin[hy * WWI + wx];
        }
    }
    const _Float16 v = (_Float16)siluf(s);
    const int lT = ww * HH + h;
    const size_t sk = (size_t)BB * LLEN * DE;
    const size_t bb = (size_t)b * LLEN * DE;
    xs[0 * sk + bb + (size_t)l * DE + d]               = v;
    xs[1 * sk + bb + (size_t)lT * DE + d]              = v;
    xs[2 * sk + bb + (size_t)(LLEN - 1 - l) * DE + d]  = v;
    xs[3 * sk + bb + (size_t)(LLEN - 1 - lT) * DE + d] = v;
}

// ---------------------------------------------------------------------------
// Selective scan: one thread per (k,b,d) chain, NS=16 states in registers.
// ys[k][b][l][d] = sum_n h_n*C_n + Ds*u
// ---------------------------------------------------------------------------
__global__ void scan_kernel(const float* __restrict__ delta,
                            const _Float16* __restrict__ u,
                            const float* __restrict__ xdbl,
                            const float* __restrict__ A_logs,
                            const float* __restrict__ Ds,
                            float* __restrict__ ys)
{
    int t = blockIdx.x * blockDim.x + threadIdx.x;
    if (t >= KDIR * BB * DE) return;
    const int d   = t % DE;
    const int rem = t / DE;
    const int b   = rem % BB;
    const int k   = rem / BB;

    float Arow[NSS];
#pragma unroll
    for (int n = 0; n < NSS; ++n)
        Arow[n] = -expf(A_logs[(size_t)(k * DE + d) * NSS + n]);
    const float Dv = Ds[k * DE + d];

    float h[NSS];
#pragma unroll
    for (int n = 0; n < NSS; ++n) h[n] = 0.0f;

    const size_t kb = (size_t)(k * BB + b) * LLEN;
    for (int l = 0; l < LLEN; ++l) {
        const size_t row = kb + l;
        __builtin_prefetch(&delta[(row + 8) * DE + d], 0, 1);
        __builtin_prefetch(&xdbl[(row + 8) * CDBL], 0, 1);
        const float dl = delta[row * DE + d];
        const float uu = (float)u[row * DE + d];
        const float x  = dl * uu;
        const float* bc = xdbl + row * CDBL;
        float y = 0.0f;
#pragma unroll
        for (int n = 0; n < NSS; ++n) {
            h[n] = expf(dl * Arow[n]) * h[n] + x * bc[RANK + n];
            y += h[n] * bc[RANK + NSS + n];
        }
        ys[row * DE + d] = y + Dv * uu;
    }
}

// ---------------------------------------------------------------------------
// Merge 4 scan directions: y[b][l][d]
// ---------------------------------------------------------------------------
__global__ void merge_kernel(const float* __restrict__ ys, float* __restrict__ ym)
{
    int idx = blockIdx.x * blockDim.x + threadIdx.x;   // M*DE
    if (idx >= MM * DE) return;
    const int d = idx % DE;
    const int m = idx / DE;
    const int b = m / LLEN;
    const int l = m % LLEN;
    const int h = l / WWI, ww = l % WWI;
    const int lT = ww * HH + h;
    const size_t sk = (size_t)BB * LLEN * DE;
    const size_t bb = (size_t)b * LLEN * DE;
    ym[idx] = ys[0 * sk + bb + (size_t)l * DE + d]
            + ys[2 * sk + bb + (size_t)(LLEN - 1 - l) * DE + d]
            + ys[1 * sk + bb + (size_t)lT * DE + d]
            + ys[3 * sk + bb + (size_t)(LLEN - 1 - lT) * DE + d];
}

// ---------------------------------------------------------------------------
// Depthwise 3x3 + BN + GELU on NHWC xn -> conv_x (NCHW)
// ---------------------------------------------------------------------------
__global__ void dwconv_bn_gelu_kernel(const float* __restrict__ xn,
                                      const float* __restrict__ w,
                                      const float* __restrict__ bias,
                                      const float* __restrict__ bg,
                                      const float* __restrict__ bb_,
                                      const float* __restrict__ bm,
                                      const float* __restrict__ bv,
                                      float* __restrict__ outNCHW)
{
    int idx = blockIdx.x * blockDim.x + threadIdx.x;   // B*C*L  (NCHW order)
    if (idx >= BB * CC * LLEN) return;
    const int l = idx % LLEN;
    const int c = (idx / LLEN) % CC;
    const int b = idx / (LLEN * CC);
    const int h = l / WWI, ww = l % WWI;
    float s = bias[c];
#pragma unroll
    for (int ky = 0; ky < 3; ++ky) {
        const int hy = h + ky - 1;
        if (hy < 0 || hy >= HH) continue;
#pragma unroll
        for (int kx = 0; kx < 3; ++kx) {
            const int wx = ww + kx - 1;
            if (wx < 0 || wx >= WWI) continue;
            s += w[c * 9 + ky * 3 + kx] * xn[((size_t)b * LLEN + hy * WWI + wx) * CC + c];
        }
    }
    const float sc = bg[c] * rsqrtf(bv[c] + 1e-5f);
    s = (s - bm[c]) * sc + bb_[c];
    outNCHW[idx] = geluf(s);
}

// ---------------------------------------------------------------------------
// Global average pool over spatial, one block per (b,c)
// ---------------------------------------------------------------------------
__global__ __launch_bounds__(256) void gap_kernel(const float* __restrict__ cx,
                                                  float* __restrict__ gap)
{
    __shared__ float sm[256];
    const int bc = blockIdx.x;
    float s = 0.0f;
    for (int i = threadIdx.x; i < LLEN; i += 256) s += cx[(size_t)bc * LLEN + i];
    sm[threadIdx.x] = s;
    __syncthreads();
    for (int o = 128; o > 0; o >>= 1) {
        if ((int)threadIdx.x < o) sm[threadIdx.x] += sm[threadIdx.x + o];
        __syncthreads();
    }
    if (threadIdx.x == 0) gap[bc] = sm[0] / (float)LLEN;
}

// ---------------------------------------------------------------------------
// Channel-interaction branch: gap -> 1x1 -> BN -> GELU -> 1x1 -> sigmoid
// One block per batch.
// ---------------------------------------------------------------------------
__global__ void ci_kernel(const float* __restrict__ gap,
                          const float* __restrict__ w1, const float* __restrict__ b1,
                          const float* __restrict__ bg, const float* __restrict__ bb_,
                          const float* __restrict__ bm, const float* __restrict__ bv,
                          const float* __restrict__ w2, const float* __restrict__ b2,
                          float* __restrict__ cmap)
{
    __shared__ float gv[CC];
    __shared__ float t1[CC / 8];
    const int b = blockIdx.x;
    const int t = threadIdx.x;
    if (t < CC) gv[t] = gap[b * CC + t];
    __syncthreads();
    if (t < CC / 8) {
        float s = b1[t];
        for (int c = 0; c < CC; ++c) s += gv[c] * w1[t * CC + c];
        const float sc = bg[t] * rsqrtf(bv[t] + 1e-5f);
        s = (s - bm[t]) * sc + bb_[t];
        t1[t] = geluf(s);
    }
    __syncthreads();
    if (t < CC) {
        float s = b2[t];
        for (int j = 0; j < CC / 8; ++j) s += t1[j] * w2[t * (CC / 8) + j];
        cmap[b * CC + t] = sigmoidf(s);
    }
}

// ---------------------------------------------------------------------------
// Spatial-interaction branch: per-pixel 96->6 (BN,GELU) ->1, sigmoid.
// ---------------------------------------------------------------------------
__global__ void si_kernel(const float* __restrict__ vss,
                          const float* __restrict__ w1, const float* __restrict__ b1,
                          const float* __restrict__ bg, const float* __restrict__ bb_,
                          const float* __restrict__ bm, const float* __restrict__ bv,
                          const float* __restrict__ w2, const float* __restrict__ b2,
                          float* __restrict__ smap)
{
    int m = blockIdx.x * blockDim.x + threadIdx.x;
    if (m >= MM) return;
    const float* xr = vss + (size_t)m * CC;
    float sm = b2[0];
#pragma unroll
    for (int j = 0; j < CC / 16; ++j) {
        float s = b1[j];
        for (int c = 0; c < CC; ++c) s += xr[c] * w1[j * CC + c];
        const float sc = bg[j] * rsqrtf(bv[j] + 1e-5f);
        s = (s - bm[j]) * sc + bb_[j];
        sm += geluf(s) * w2[j];
    }
    smap[m] = sigmoidf(sm);
}

// ---------------------------------------------------------------------------
// y = vss*sigmoid(channel_map) + sigmoid(spatial_map)*conv_x  -> f16
// ---------------------------------------------------------------------------
__global__ void combine_kernel(const float* __restrict__ vss,
                               const float* __restrict__ cmap,
                               const float* __restrict__ smap,
                               const float* __restrict__ convx,
                               _Float16* __restrict__ a_y)
{
    int idx = blockIdx.x * blockDim.x + threadIdx.x;   // M*C
    if (idx >= MM * CC) return;
    const int c = idx % CC;
    const int m = idx / CC;
    const int b = m / LLEN;
    const int l = m % LLEN;
    const float v = vss[idx] * cmap[b * CC + c]
                  + smap[m] * convx[((size_t)b * CC + c) * LLEN + l];
    a_y[idx] = (_Float16)v;
}

// ---------------------------------------------------------------------------
// Depthwise 3x3 + bias on NHWC [B,L,48] (SGFN branch)
// ---------------------------------------------------------------------------
__global__ void dwconv_sg_kernel(const float* __restrict__ x2n,
                                 const float* __restrict__ w,
                                 const float* __restrict__ bias,
                                 float* __restrict__ x2c)
{
    int idx = blockIdx.x * blockDim.x + threadIdx.x;   // B*L*48
    if (idx >= BB * LLEN * CH2) return;
    const int c = idx % CH2;
    const int l = (idx / CH2) % LLEN;
    const int b = idx / (CH2 * LLEN);
    const int h = l / WWI, ww = l % WWI;
    float s = bias[c];
#pragma unroll
    for (int ky = 0; ky < 3; ++ky) {
        const int hy = h + ky - 1;
        if (hy < 0 || hy >= HH) continue;
#pragma unroll
        for (int kx = 0; kx < 3; ++kx) {
            const int wx = ww + kx - 1;
            if (wx < 0 || wx >= WWI) continue;
            s += w[c * 9 + ky * 3 + kx] * x2n[((size_t)b * LLEN + hy * WWI + wx) * CH2 + c];
        }
    }
    x2c[idx] = s;
}

// ---------------------------------------------------------------------------
// a2[m, 0..63] = f16( x1[m,c] * x2c[m,c] ) for c<48, 0 for 48..63 (K padding)
// ---------------------------------------------------------------------------
__global__ void mul_to_h_kernel(const float* __restrict__ tfc1,
                                const float* __restrict__ x2c,
                                _Float16* __restrict__ a2)
{
    int idx = blockIdx.x * blockDim.x + threadIdx.x;   // M*64
    if (idx >= MM * CH2_PAD) return;
    const int c = idx % CH2_PAD;
    const int m = idx / CH2_PAD;
    _Float16 v = (_Float16)0.0f;
    if (c < CH2)
        v = (_Float16)(tfc1[(size_t)m * CC + c] * x2c[(size_t)m * CH2 + c]);
    a2[idx] = v;
}

// ---------------------------------------------------------------------------
// Host orchestration
// ---------------------------------------------------------------------------
extern "C" void kernel_launch(void* const* d_in, const int* in_sizes, int n_in,
                              void* d_out, int out_size, void* d_ws, size_t ws_size,
                              hipStream_t stream)
{
    (void)in_sizes; (void)n_in; (void)out_size; (void)ws_size;

    // ---- inputs (setup_inputs dict order) ----
    const float* x         = (const float*)d_in[0];
    const float* norm_g    = (const float*)d_in[1];
    const float* norm_b    = (const float*)d_in[2];
    const float* in_proj_w = (const float*)d_in[3];
    const float* conv_w    = (const float*)d_in[4];
    const float* conv_b    = (const float*)d_in[5];
    const float* x_proj_w  = (const float*)d_in[6];
    const float* dt_w      = (const float*)d_in[7];
    const float* dt_b      = (const float*)d_in[8];
    const float* A_logs    = (const float*)d_in[9];
    const float* Ds        = (const float*)d_in[10];
    const float* onorm_g   = (const float*)d_in[11];
    const float* onorm_b   = (const float*)d_in[12];
    const float* out_proj_w= (const float*)d_in[13];
    const float* dw_w      = (const float*)d_in[14];
    const float* dw_b      = (const float*)d_in[15];
    const float* dwbn_g    = (const float*)d_in[16];
    const float* dwbn_b    = (const float*)d_in[17];
    const float* dwbn_m    = (const float*)d_in[18];
    const float* dwbn_v    = (const float*)d_in[19];
    const float* ci1_w     = (const float*)d_in[20];
    const float* ci1_b     = (const float*)d_in[21];
    const float* cibn_g    = (const float*)d_in[22];
    const float* cibn_b    = (const float*)d_in[23];
    const float* cibn_m    = (const float*)d_in[24];
    const float* cibn_v    = (const float*)d_in[25];
    const float* ci2_w     = (const float*)d_in[26];
    const float* ci2_b     = (const float*)d_in[27];
    const float* si1_w     = (const float*)d_in[28];
    const float* si1_b     = (const float*)d_in[29];
    const float* sibn_g    = (const float*)d_in[30];
    const float* sibn_b    = (const float*)d_in[31];
    const float* sibn_m    = (const float*)d_in[32];
    const float* sibn_v    = (const float*)d_in[33];
    const float* si2_w     = (const float*)d_in[34];
    const float* si2_b     = (const float*)d_in[35];
    const float* proj_w    = (const float*)d_in[36];
    const float* proj_b    = (const float*)d_in[37];
    const float* norm3_g   = (const float*)d_in[38];
    const float* norm3_b   = (const float*)d_in[39];
    const float* fc1_w     = (const float*)d_in[40];
    const float* fc1_b     = (const float*)d_in[41];
    const float* sgln_g    = (const float*)d_in[42];
    const float* sgln_b    = (const float*)d_in[43];
    const float* sg_conv_w = (const float*)d_in[44];
    const float* sg_conv_b = (const float*)d_in[45];
    const float* fc2_w     = (const float*)d_in[46];
    const float* fc2_b     = (const float*)d_in[47];

    // ---- workspace bump allocator ----
    char* base = (char*)d_ws;
    size_t off = 0;
    auto alloc = [&](size_t bytes) -> void* {
        void* p = base + off;
        off = (off + bytes + 255) & ~(size_t)255;
        return p;
    };

    float*     xn      = (float*)    alloc((size_t)MM * CC * 4);
    _Float16*  xn_h    = (_Float16*) alloc((size_t)MM * CC * 2);
    _Float16*  w_inp   = (_Float16*) alloc((size_t)D2 * CC * 2);
    _Float16*  w_xprj  = (_Float16*) alloc((size_t)KDIR * CDBL_PAD * DE * 2);
    _Float16*  w_dt    = (_Float16*) alloc((size_t)KDIR * DE * RANK_PAD * 2);
    _Float16*  w_oprj  = (_Float16*) alloc((size_t)CC * DE * 2);
    _Float16*  w_prj   = (_Float16*) alloc((size_t)CC * CC * 2);
    _Float16*  w_f1    = (_Float16*) alloc((size_t)CC * CC * 2);
    _Float16*  w_f2    = (_Float16*) alloc((size_t)CC * CH2_PAD * 2);
    float*     xz      = (float*)    alloc((size_t)MM * D2 * 4);
    float*     xx      = (float*)    alloc((size_t)BB * DE * LLEN * 4);
    float*     zbuf    = (float*)    alloc((size_t)MM * DE * 4);
    _Float16*  xs      = (_Float16*) alloc((size_t)KDIR * BB * LLEN * DE * 2);
    float*     xdbl    = (float*)    alloc((size_t)KDIR * BB * LLEN * CDBL * 4);
    _Float16*  dts_h   = (_Float16*) alloc((size_t)KDIR * BB * LLEN * RANK_PAD * 2);
    float*     delta   = (float*)    alloc((size_t)KDIR * BB * LLEN * DE * 4);
    float*     ysb     = (float*)    alloc((size_t)KDIR * BB * LLEN * DE * 4);
    float*     ym      = (float*)    alloc((size_t)MM * DE * 4);
    _Float16*  a_out   = (_Float16*) alloc((size_t)MM * DE * 2);
    float*     vss     = (float*)    alloc((size_t)MM * CC * 4);
    float*     convx   = (float*)    alloc((size_t)BB * CC * LLEN * 4);
    float*     gap     = (float*)    alloc((size_t)BB * CC * 4);
    float*     cmap    = (float*)    alloc((size_t)BB * CC * 4);
    float*     smap    = (float*)    alloc((size_t)MM * 4);
    _Float16*  a_y     = (_Float16*) alloc((size_t)MM * CC * 2);
    float*     ybuf    = (float*)    alloc((size_t)MM * CC * 4);
    float*     y3      = (float*)    alloc((size_t)MM * CC * 4);
    _Float16*  y3h     = (_Float16*) alloc((size_t)MM * CC * 2);
    float*     tfc1    = (float*)    alloc((size_t)MM * CC * 4);
    float*     x2n     = (float*)    alloc((size_t)MM * CH2 * 4);
    float*     x2c     = (float*)    alloc((size_t)MM * CH2 * 4);
    _Float16*  a2      = (_Float16*) alloc((size_t)MM * CH2_PAD * 2);

    auto cvt = [&](const float* s, _Float16* d, int n) {
        cvt_f32_f16_kernel<<<(n + 255) / 256, 256, 0, stream>>>(s, d, n);
    };
    auto cvt_pad = [&](const float* s, int ld_src, int coff, _Float16* d, int ld_dst,
                       int rows_src, int cols_src, int rows_pad) {
        int total = rows_pad * ld_dst;
        cvt_pad_kernel<<<(total + 255) / 256, 256, 0, stream>>>(
            s, ld_src, coff, d, ld_dst, rows_src, cols_src, total);
    };
    auto gemm = [&](const _Float16* Ap, const _Float16* Wp, float* Cp,
                    int Mr, int Nc, int Kd, int lda, int ldb, int ldc,
                    const float* bias, const float* resid, int ldr, int act) {
        int tiles = (Mr >> 4) * ((Nc + 15) >> 4);
        int blocks = (tiles + 7) / 8;
        switch (Kd) {
        case 32:
            gemm_wmma_kernel<32><<<blocks, 256, 0, stream>>>(
                Ap, Wp, Cp, Mr, Nc, lda, ldb, ldc, bias, resid, ldr, act); break;
        case 64:
            gemm_wmma_kernel<64><<<blocks, 256, 0, stream>>>(
                Ap, Wp, Cp, Mr, Nc, lda, ldb, ldc, bias, resid, ldr, act); break;
        case 96:
            gemm_wmma_kernel<96><<<blocks, 256, 0, stream>>>(
                Ap, Wp, Cp, Mr, Nc, lda, ldb, ldc, bias, resid, ldr, act); break;
        default:
            gemm_wmma_kernel<192><<<blocks, 256, 0, stream>>>(
                Ap, Wp, Cp, Mr, Nc, lda, ldb, ldc, bias, resid, ldr, act); break;
        }
    };

    // ---- weight conversions (f32 -> f16, padded where needed) ----
    cvt(in_proj_w,  w_inp,  D2 * CC);
    cvt(out_proj_w, w_oprj, CC * DE);
    cvt(proj_w,     w_prj,  CC * CC);
    cvt(fc1_w,      w_f1,   CC * CC);
    for (int k = 0; k < KDIR; ++k) {
        // x_proj_w[k]: [38,192] -> padded [48,192]
        cvt_pad(x_proj_w + (size_t)k * CDBL * DE, DE, 0,
                w_xprj + (size_t)k * CDBL_PAD * DE, DE, CDBL, DE, CDBL_PAD);
        // dt_w[k]: [192,6] -> padded [192,32]
        cvt_pad(dt_w + (size_t)k * DE * RANK, RANK, 0,
                w_dt + (size_t)k * DE * RANK_PAD, RANK_PAD, DE, RANK, DE);
    }
    // fc2_w: [96,48] -> padded [96,64]
    cvt_pad(fc2_w, CH2, 0, w_f2, CH2_PAD, CC, CH2, CC);

    // ---- 1) norm1 ----
    ln_kernel<<<MM / 8, 256, 0, stream>>>(x, CC, 0, norm_g, norm_b, nullptr,
                                          xn, xn_h, MM, CC, 1e-6f);

    // ---- 2) in_proj GEMM: [M,96] x [384,96]^T -> [M,384] ----
    gemm(xn_h, w_inp, xz, MM, D2, CC, CC, CC, D2, nullptr, nullptr, 0, ACT_NONE);

    // ---- 3) split + silu(z) ----
    split_inproj_kernel<<<(MM * DE + 255) / 256, 256, 0, stream>>>(xz, xx, zbuf);

    // ---- 4) dwconv + silu -> xs (4 directions, f16) ----
    dwconv_xs_kernel<<<(BB * DE * LLEN + 255) / 256, 256, 0, stream>>>(xx, conv_w, conv_b, xs);

    // ---- 5) x_proj GEMMs (per direction): [B*L,192] x [48,192]^T -> [B*L,38] ----
    for (int k = 0; k < KDIR; ++k)
        gemm(xs + (size_t)k * BB * LLEN * DE, w_xprj + (size_t)k * CDBL_PAD * DE,
             xdbl + (size_t)k * BB * LLEN * CDBL,
             BB * LLEN, CDBL, DE, DE, DE, CDBL, nullptr, nullptr, 0, ACT_NONE);

    // ---- 6) dts slice -> f16 padded [rows, 32]; dt GEMMs + softplus -> delta ----
    cvt_pad(xdbl, CDBL, 0, dts_h, RANK_PAD, KDIR * BB * LLEN, RANK, KDIR * BB * LLEN);
    for (int k = 0; k < KDIR; ++k)
        gemm(dts_h + (size_t)k * BB * LLEN * RANK_PAD, w_dt + (size_t)k * DE * RANK_PAD,
             delta + (size_t)k * BB * LLEN * DE,
             BB * LLEN, DE, RANK_PAD, RANK_PAD, RANK_PAD, DE,
             dt_b + (size_t)k * DE, nullptr, 0, ACT_SOFTPLUS);

    // ---- 7) selective scan ----
    scan_kernel<<<(KDIR * BB * DE + 191) / 192, 192, 0, stream>>>(
        delta, xs, xdbl, A_logs, Ds, ysb);

    // ---- 8) merge directions ----
    merge_kernel<<<(MM * DE + 255) / 256, 256, 0, stream>>>(ysb, ym);

    // ---- 9) out_norm LN * z -> f16 A ----
    ln_kernel<<<MM / 8, 256, 0, stream>>>(ym, DE, 0, onorm_g, onorm_b, zbuf,
                                          nullptr, a_out, MM, DE, 1e-5f);

    // ---- 10) out_proj GEMM: [M,192] x [96,192]^T -> vss [M,96] ----
    gemm(a_out, w_oprj, vss, MM, CC, DE, DE, DE, CC, nullptr, nullptr, 0, ACT_NONE);

    // ---- 11) conv branch ----
    dwconv_bn_gelu_kernel<<<(BB * CC * LLEN + 255) / 256, 256, 0, stream>>>(
        xn, dw_w, dw_b, dwbn_g, dwbn_b, dwbn_m, dwbn_v, convx);
    gap_kernel<<<BB * CC, 256, 0, stream>>>(convx, gap);
    ci_kernel<<<BB, 128, 0, stream>>>(gap, ci1_w, ci1_b, cibn_g, cibn_b, cibn_m,
                                      cibn_v, ci2_w, ci2_b, cmap);
    si_kernel<<<(MM + 127) / 128, 128, 0, stream>>>(vss, si1_w, si1_b, sibn_g,
                                                    sibn_b, sibn_m, sibn_v,
                                                    si2_w, si2_b, smap);

    // ---- 12) gated combine -> f16 ----
    combine_kernel<<<(MM * CC + 255) / 256, 256, 0, stream>>>(vss, cmap, smap, convx, a_y);

    // ---- 13) proj GEMM + bias + residual(xn) ----
    gemm(a_y, w_prj, ybuf, MM, CC, CC, CC, CC, CC, proj_b, xn, CC, ACT_NONE);

    // ---- 14) norm3 ----
    ln_kernel<<<MM / 8, 256, 0, stream>>>(ybuf, CC, 0, norm3_g, norm3_b, nullptr,
                                          y3, y3h, MM, CC, 1e-6f);

    // ---- 15) SGFN ----
    gemm(y3h, w_f1, tfc1, MM, CC, CC, CC, CC, CC, fc1_b, nullptr, 0, ACT_GELU);
    ln_kernel<<<MM / 8, 256, 0, stream>>>(tfc1, CC, CH2, sgln_g, sgln_b, nullptr,
                                          x2n, nullptr, MM, CH2, 1e-5f);
    dwconv_sg_kernel<<<(BB * LLEN * CH2 + 255) / 256, 256, 0, stream>>>(
        x2n, sg_conv_w, sg_conv_b, x2c);
    mul_to_h_kernel<<<(MM * CH2_PAD + 255) / 256, 256, 0, stream>>>(tfc1, x2c, a2);

    // ---- 16) fc2 GEMM + bias + residual(y3) -> d_out ----
    gemm(a2, w_f2, (float*)d_out, MM, CC, CH2_PAD, CH2_PAD, CH2_PAD, CC,
         fc2_b, y3, CC, ACT_NONE);
}
